// LSTMAutoencoder_56315611185744
// MI455X (gfx1250) — matrix operand order repaired and checked by reading, not compiled, not run on hardware
//
#include <hip/hip_runtime.h>
#include <hip/hip_bf16.h>

#define BDIM 128
#define TDIM 512
#define IDIM 128
#define HDIM 512
#define GDIM 2048

typedef __attribute__((ext_vector_type(16))) __bf16 v16bf;
typedef __attribute__((ext_vector_type(8)))  __bf16 v8bf;
typedef __attribute__((ext_vector_type(8)))  float  v8f;

// ---------------------------------------------------------------------------
// Fragment loaders for V_WMMA_F32_16X16X32_BF16 (wave32).
// A (16x32, MxK): lanes 0-15 -> M=lane, K = {0..7, 16..23}; lanes 16-31 -> +8.
// B (32x16, KxN): lane -> N = lane&15, K = {0..7, 16..23}; lanes 16-31 -> +8.
// prow = matrix_base + row * ld  (precomputed per wave, hoisted out of K loop).
// ---------------------------------------------------------------------------
__device__ __forceinline__ v16bf load_a_frag(const __bf16* __restrict__ prow,
                                             int k0, int lane) {
  const __bf16* p = prow + k0 + ((lane & 16) ? 8 : 0);
  v8bf lo = *(const v8bf*)(p);        // K = kb+0..7
  v8bf hi = *(const v8bf*)(p + 16);   // K = kb+16..23
  v16bf a;
#pragma unroll
  for (int i = 0; i < 8; ++i) { a[i] = lo[i]; a[i + 8] = hi[i]; }
  return a;
}

__device__ __forceinline__ v16bf load_b_frag(const __bf16* __restrict__ prow,
                                             int k0, int lane) {
  const __bf16* p = prow + k0 + ((lane & 16) ? 8 : 0);
  v8bf lo = *(const v8bf*)(p);        // K = kb+0..7
  v8bf hi = *(const v8bf*)(p + 16);   // K = kb+16..23
  v16bf b;
#pragma unroll
  for (int i = 0; i < 8; ++i) { b[i] = lo[i]; b[i + 8] = hi[i]; }
  return b;
}

__device__ __forceinline__ float sigmoidf_fast(float x) {
  return 1.0f / (1.0f + __expf(-x));
}

// ---------------------------------------------------------------------------
// One LSTM timestep, fully fused:
//   g = [x_t | h_prev] @ [Wih ; Whh]^T + bias ;  gates ; c,h update.
// Grid: 64 WGs x 256 thr. WG = 32 batch rows x 32 h-columns.
// 8 waves = 2 row-halves x 4 gates; each wave owns one gate's full 32
// columns (two 16x16 N-tiles, two accumulators, shared A fragment).
// K loop split into x-part and h-part: no per-iteration pointer selects.
// ---------------------------------------------------------------------------
__global__ __launch_bounds__(256) void lstm_step_kernel(
    const __bf16* __restrict__ xt, int ldx, int kin,
    const __bf16* __restrict__ Wih,   // [GDIM, kin]  row-major
    const __bf16* __restrict__ Whh,   // [GDIM, HDIM] row-major
    const float*  __restrict__ bias,  // [GDIM]
    const __bf16* __restrict__ h_in,  // [BDIM, HDIM]
    __bf16*       __restrict__ h_out, // [BDIM, HDIM]
    float*        __restrict__ cst,   // [BDIM, HDIM]
    __bf16*       __restrict__ seq_out, // [BDIM, TDIM, HDIM] or nullptr
    int t) {
  __shared__ float lds[16 * 256];     // 16 tiles x 16x16 f32 = 16 KB

  const int tid  = threadIdx.x;
  const int wave = tid >> 5;
  const int lane = tid & 31;
  const int mblk = blockIdx.x & 3;    // 4 blocks of 32 batch rows
  const int hgrp = blockIdx.x >> 2;   // 16 groups of 32 h-columns
  const int gate  = wave & 3;
  const int mhalf = wave >> 2;

  const int rowA  = mblk * 32 + mhalf * 16 + (lane & 15);
  const int nBase = gate * HDIM + hgrp * 32;            // W row of tile 0
  const int nRow0 = nBase + (lane & 15);
  const int nRow1 = nBase + 16 + (lane & 15);

  v8f acc0 = {};
  v8f acc1 = {};

  // --- x-part: g += x_t @ Wih^T ---
  {
    const __bf16* pa  = xt  + (size_t)rowA  * ldx;
    const __bf16* pb0 = Wih + (size_t)nRow0 * kin;
    const __bf16* pb1 = Wih + (size_t)nRow1 * kin;
#pragma unroll 4
    for (int k0 = 0; k0 < kin; k0 += 32) {
      v16bf a  = load_a_frag(pa,  k0, lane);
      v16bf b0 = load_b_frag(pb0, k0, lane);
      v16bf b1 = load_b_frag(pb1, k0, lane);
      acc0 = __builtin_amdgcn_wmma_f32_16x16x32_bf16(false, a, false, b0,
                                                     (short)0, acc0, false, false);
      acc1 = __builtin_amdgcn_wmma_f32_16x16x32_bf16(false, a, false, b1,
                                                     (short)0, acc1, false, false);
    }
  }
  // --- h-part: g += h_prev @ Whh^T ---
  {
    const __bf16* pa  = h_in + (size_t)rowA  * HDIM;
    const __bf16* pb0 = Whh  + (size_t)nRow0 * HDIM;
    const __bf16* pb1 = Whh  + (size_t)nRow1 * HDIM;
#pragma unroll 4
    for (int k0 = 0; k0 < HDIM; k0 += 32) {
      v16bf a  = load_a_frag(pa,  k0, lane);
      v16bf b0 = load_b_frag(pb0, k0, lane);
      v16bf b1 = load_b_frag(pb1, k0, lane);
      acc0 = __builtin_amdgcn_wmma_f32_16x16x32_bf16(false, a, false, b0,
                                                     (short)0, acc0, false, false);
      acc1 = __builtin_amdgcn_wmma_f32_16x16x32_bf16(false, a, false, b1,
                                                     (short)0, acc1, false, false);
    }
  }

  // Stash both tiles. Tile slot = wave*2 + half. Element r of C/D:
  // M = (lane>>4)*8 + r, N = lane&15.
#pragma unroll
  for (int r = 0; r < 8; ++r) {
    const int off = (((lane >> 4) * 8 + r) << 4) + (lane & 15);
    lds[(wave * 2 + 0) * 256 + off] = acc0[r];
    lds[(wave * 2 + 1) * 256 + off] = acc1[r];
  }
  __syncthreads();

  // Fused gate math: 1024 outputs (32 rows x 32 h-cols), 4 per thread.
  for (int e = tid; e < 1024; e += 256) {
    const int m   = e >> 5;            // 0..31
    const int col = e & 31;            // 0..31
    const int p   = col >> 4;          // N-tile half
    const int nn  = col & 15;
    const int mh  = m >> 4;            // row half -> wave group
    const int off = ((m & 15) << 4) + nn;
    const int b    = mblk * 32 + m;
    const int hcol = hgrp * 32 + col;

    float xi = lds[(((mh * 4 + 0) * 2 + p)) * 256 + off] + bias[0 * HDIM + hcol];
    float xf = lds[(((mh * 4 + 1) * 2 + p)) * 256 + off] + bias[1 * HDIM + hcol];
    float xg = lds[(((mh * 4 + 2) * 2 + p)) * 256 + off] + bias[2 * HDIM + hcol];
    float xo = lds[(((mh * 4 + 3) * 2 + p)) * 256 + off] + bias[3 * HDIM + hcol];

    float ig = sigmoidf_fast(xi);
    float fg = sigmoidf_fast(xf);
    float gg = tanhf(xg);
    float og = sigmoidf_fast(xo);

    const size_t ci = (size_t)b * HDIM + hcol;
    float cn = fg * cst[ci] + ig * gg;
    cst[ci] = cn;
    float hn = og * tanhf(cn);
    __bf16 hb = (__bf16)hn;
    h_out[ci] = hb;
    if (seq_out)
      seq_out[(size_t)b * TDIM * HDIM + (size_t)t * HDIM + hcol] = hb;
  }
}

// ---------------------------------------------------------------------------
// Output projection: out[M, I] = A[M, H] @ Wout^T + bout  (M = B*T), f32 out.
// One wave per 16x16 tile, 8 waves / WG.
// ---------------------------------------------------------------------------
__global__ __launch_bounds__(256) void proj_kernel(
    const __bf16* __restrict__ A,     // [B*T, HDIM]
    const __bf16* __restrict__ W,     // [IDIM, HDIM]
    const float*  __restrict__ bias,  // [IDIM]
    float*        __restrict__ out) { // [B*T, IDIM]
  const int lane = threadIdx.x & 31;
  const int wave = threadIdx.x >> 5;
  const int tile = blockIdx.x * 8 + wave;
  const int NT = IDIM / 16;           // 8
  const int mt = tile / NT;
  const int nt = tile % NT;

  const __bf16* pa = A + (size_t)(mt * 16 + (lane & 15)) * HDIM;
  const __bf16* pb = W + (size_t)(nt * 16 + (lane & 15)) * HDIM;

  v8f acc = {};
#pragma unroll 4
  for (int k0 = 0; k0 < HDIM; k0 += 32) {
    v16bf a = load_a_frag(pa, k0, lane);
    v16bf b = load_b_frag(pb, k0, lane);
    acc = __builtin_amdgcn_wmma_f32_16x16x32_bf16(false, a, false, b,
                                                  (short)0, acc, false, false);
  }
#pragma unroll
  for (int r = 0; r < 8; ++r) {
    const int m = mt * 16 + (lane >> 4) * 8 + r;
    const int n = nt * 16 + (lane & 15);
    out[(size_t)m * IDIM + n] = acc[r] + bias[n];
  }
}

// ---------------------------------------------------------------------------
// Helpers
// ---------------------------------------------------------------------------
__global__ void cvt_bf16_kernel(const float* __restrict__ s,
                                __bf16* __restrict__ d, int n) {
  int i = blockIdx.x * blockDim.x + threadIdx.x;
  if (i < n) d[i] = (__bf16)s[i];
}

__global__ void zero_state_kernel(__bf16* __restrict__ h,
                                  float* __restrict__ c, int n) {
  int i = blockIdx.x * blockDim.x + threadIdx.x;
  if (i < n) { h[i] = (__bf16)0.0f; c[i] = 0.0f; }
}

__global__ void copy_bf16_kernel(const __bf16* __restrict__ s,
                                 __bf16* __restrict__ d, int n) {
  int i = blockIdx.x * blockDim.x + threadIdx.x;
  if (i < n) d[i] = s[i];
}

// ---------------------------------------------------------------------------
extern "C" void kernel_launch(void* const* d_in, const int* in_sizes, int n_in,
                              void* d_out, int out_size, void* d_ws, size_t ws_size,
                              hipStream_t stream) {
  (void)in_sizes; (void)n_in; (void)out_size; (void)ws_size;

  const float* x        = (const float*)d_in[0];
  const float* enc_Wih0 = (const float*)d_in[1];
  const float* enc_Whh0 = (const float*)d_in[2];
  const float* enc_b0   = (const float*)d_in[3];
  const float* enc_Wih1 = (const float*)d_in[4];
  const float* enc_Whh1 = (const float*)d_in[5];
  const float* enc_b1   = (const float*)d_in[6];
  const float* dec_Wih0 = (const float*)d_in[7];
  const float* dec_Whh0 = (const float*)d_in[8];
  const float* dec_b0   = (const float*)d_in[9];
  const float* dec_Wih1 = (const float*)d_in[10];
  const float* dec_Whh1 = (const float*)d_in[11];
  const float* dec_b1   = (const float*)d_in[12];
  const float* Wout     = (const float*)d_in[13];
  const float* bout     = (const float*)d_in[14];

  // Workspace carve (256B aligned slices)
  char* ws = (char*)d_ws;
  size_t off = 0;
  auto carve = [&](size_t bytes) -> void* {
    off = (off + 255) & ~(size_t)255;
    void* p = ws + off;
    off += bytes;
    return p;
  };
  __bf16* x_bf  = (__bf16*)carve((size_t)BDIM * TDIM * IDIM * 2);
  __bf16* wih0  = (__bf16*)carve((size_t)GDIM * IDIM * 2);
  __bf16* whh0  = (__bf16*)carve((size_t)GDIM * HDIM * 2);
  __bf16* wih1  = (__bf16*)carve((size_t)GDIM * HDIM * 2);
  __bf16* whh1  = (__bf16*)carve((size_t)GDIM * HDIM * 2);
  __bf16* dwih0 = (__bf16*)carve((size_t)GDIM * HDIM * 2);
  __bf16* dwhh0 = (__bf16*)carve((size_t)GDIM * HDIM * 2);
  __bf16* dwih1 = (__bf16*)carve((size_t)GDIM * HDIM * 2);
  __bf16* dwhh1 = (__bf16*)carve((size_t)GDIM * HDIM * 2);
  __bf16* wout  = (__bf16*)carve((size_t)IDIM * HDIM * 2);
  __bf16* seq0  = (__bf16*)carve((size_t)BDIM * TDIM * HDIM * 2);
  __bf16* seq1  = (__bf16*)carve((size_t)BDIM * TDIM * HDIM * 2);
  __bf16* hbuf0 = (__bf16*)carve((size_t)BDIM * HDIM * 2);
  __bf16* hbuf1 = (__bf16*)carve((size_t)BDIM * HDIM * 2);
  float*  cbuf  = (float*) carve((size_t)BDIM * HDIM * 4);
  __bf16* hT    = (__bf16*)carve((size_t)BDIM * HDIM * 2);

  auto cvt = [&](const float* s, __bf16* d, int n) {
    cvt_bf16_kernel<<<(n + 255) / 256, 256, 0, stream>>>(s, d, n);
  };
  cvt(x, x_bf, BDIM * TDIM * IDIM);
  cvt(enc_Wih0, wih0, GDIM * IDIM);
  cvt(enc_Whh0, whh0, GDIM * HDIM);
  cvt(enc_Wih1, wih1, GDIM * HDIM);
  cvt(enc_Whh1, whh1, GDIM * HDIM);
  cvt(dec_Wih0, dwih0, GDIM * HDIM);
  cvt(dec_Whh0, dwhh0, GDIM * HDIM);
  cvt(dec_Wih1, dwih1, GDIM * HDIM);
  cvt(dec_Whh1, dwhh1, GDIM * HDIM);
  cvt(Wout, wout, IDIM * HDIM);

  // Run one LSTM layer: xbase advances tstride elements per timestep
  // (tstride=0 => constant input, used for the decoder's repeated hT).
  auto run_layer = [&](const __bf16* xbase, int ldx, int kin, int tstride,
                       const __bf16* Wih, const __bf16* Whh, const float* bias,
                       __bf16* seq_out) {
    zero_state_kernel<<<(BDIM * HDIM + 255) / 256, 256, 0, stream>>>(
        hbuf0, cbuf, BDIM * HDIM);
    for (int t = 0; t < TDIM; ++t) {
      const __bf16* hin = (t & 1) ? hbuf1 : hbuf0;
      __bf16*       hout = (t & 1) ? hbuf0 : hbuf1;
      lstm_step_kernel<<<64, 256, 0, stream>>>(
          xbase + (size_t)tstride * t, ldx, kin, Wih, Whh, bias,
          hin, hout, cbuf, seq_out, t);
    }
    // TDIM even -> final hidden state lives in hbuf0
  };

  // Encoder
  run_layer(x_bf, TDIM * IDIM, IDIM, IDIM, wih0, whh0, enc_b0, seq0);
  run_layer(seq0, TDIM * HDIM, HDIM, HDIM, wih1, whh1, enc_b1, nullptr);
  copy_bf16_kernel<<<(BDIM * HDIM + 255) / 256, 256, 0, stream>>>(
      hbuf0, hT, BDIM * HDIM);

  // Decoder (input = hT repeated across time; tstride = 0)
  run_layer(hT, HDIM, HDIM, 0, dwih0, dwhh0, dec_b0, seq1);
  run_layer(seq1, TDIM * HDIM, HDIM, HDIM, dwih1, dwhh1, dec_b1, seq0);

  // Output projection -> d_out (f32, [B, T, I])
  const int mtiles = (BDIM * TDIM) / 16;          // 4096
  const int total_tiles = mtiles * (IDIM / 16);   // 32768
  proj_kernel<<<total_tiles / 8, 256, 0, stream>>>(seq0, wout, bout,
                                                   (float*)d_out);
}
